// FAN_39273180955145
// MI455X (gfx1250) — matrix operand order repaired and checked by reading, not compiled, not run on hardware
//
#include <hip/hip_runtime.h>
#include <hip/hip_bf16.h>
#include <math.h>

// ---------------------------------------------------------------------------
// Fractional attention net on MI455X (gfx1250), wave32 + WMMA bf16.
//
// ~260 GFLOP total, GEMM-dominated -> v_wmma_f32_16x16x32_bf16 everywhere.
// One workgroup per batch element keeps h/K/V^T resident in LDS (bf16,
// 64KB each) using CDNA5's 320KB-per-WGP LDS; Q streamed in 16-row tiles.
// 8 waves/workgroup = 2 waves/SIMD32 to hide WMMA + LDS latency.
// Kernel 2 (32896-deep GEMM) uses a pre-converted bf16 weight image and a
// double-buffered LDS slab; the raw-copy staging uses the CDNA5 async
// global->LDS path (ASYNCcnt + s_wait_asynccnt).
// ---------------------------------------------------------------------------

#define BATCH 2048
#define SEQ   256
#define DIM   128
#define NSC   16
#define NLAY  2
#define COMBW (DIM * (SEQ + 1))   // 32896
#define TPB1  256                 // 8 waves in kernel 1

typedef float  v8f   __attribute__((ext_vector_type(8)));
typedef __bf16 v16bf __attribute__((ext_vector_type(16)));
typedef int    v4i_vs __attribute__((vector_size(16)));   // builtin's pointee

#if defined(__has_builtin)
#if __has_builtin(__builtin_amdgcn_global_load_async_to_lds_b128)
#define HAVE_ASYNC_LDS 1
#endif
#endif

__device__ __forceinline__ v8f wmma_bf16(v16bf a, v16bf b, v8f c) {
  return __builtin_amdgcn_wmma_f32_16x16x32_bf16(false, a, false, b,
                                                 (short)0, c, false, false);
}

// A fragment (16x32 bf16, M x K) from a row-major [row][k] buffer.
// ISA layout: lane&15 = M row; lanes 0-15 hold K = k0+0..7 and k0+16..23,
// lanes 16-31 hold K = k0+8..15 and k0+24..31 -> two 16B contiguous loads.
__device__ __forceinline__ v16bf load_A(const __bf16* p, int ld, int row0,
                                        int k0, int lane) {
  int r  = row0 + (lane & 15);
  int kk = k0 + ((lane >> 4) << 3);
  const __bf16* q = p + r * ld + kk;
  v16bf a;
  ((uint4*)&a)[0] = *(const uint4*)(q);
  ((uint4*)&a)[1] = *(const uint4*)(q + 16);
  return a;
}

// B fragment (32x16 bf16, K x N) from an N-major [n][k] buffer
// (element B[k][n] stored at p[n*ld + k]).
// ISA layout: lane&15 = N col; lanes 0-15 hold K = k0+0..7 and k0+16..23,
// lanes 16-31 hold K = k0+8..15 and k0+24..31 -> two 16B contiguous loads.
__device__ __forceinline__ v16bf load_B(const __bf16* p, int ld, int n0,
                                        int k0, int lane) {
  int n  = n0 + (lane & 15);
  int kk = k0 + ((lane >> 4) << 3);
  const __bf16* q = p + n * ld + kk;
  v16bf b;
  ((uint4*)&b)[0] = *(const uint4*)(q);
  ((uint4*)&b)[1] = *(const uint4*)(q + 16);
  return b;
}

// ---------------------------------------------------------------------------
// Kernel 0: log fractional kernel, 256x256 f32 into workspace.
// ---------------------------------------------------------------------------
__global__ void fan_logkern_kernel(float* __restrict__ lk) {
  int s = threadIdx.x;
  if (s >= SEQ) return;
  float sum = 0.f;
  for (int t = 0; t < SEQ; ++t) {
    float td = fabsf((float)(s - t)) + 1e-9f;
    sum += powf(td, -1.4f);               // HURST - 1.5 = -1.4
  }
  float inv = 1.f / sum;
  for (int t = 0; t < SEQ; ++t) {
    float td = fabsf((float)(s - t)) + 1e-9f;
    lk[s * SEQ + t] = logf(powf(td, -1.4f) * inv + 1e-9f);
  }
}

// ---------------------------------------------------------------------------
// Kernel 0b: pre-convert h1_W (32896x128 f32, row-major) into an N-major
// bf16 image wt[n][k] so kernel 2's LDS staging is a raw b128 copy.
// ---------------------------------------------------------------------------
__global__ void fan_wconv_kernel(const float* __restrict__ h1_W,
                                 __bf16* __restrict__ wt) {
  int n = blockIdx.y;
  int k = blockIdx.x * 256 + threadIdx.x;
  if (k < COMBW) wt[(size_t)n * COMBW + k] = (__bf16)h1_W[(size_t)k * DIM + n];
}

// ---------------------------------------------------------------------------
// Kernel 1: fused embed + 2x (QKV, fractional attention, LN) per batch elem.
// One workgroup per b, 256 threads = 8 waves, ~247KB LDS.
// ---------------------------------------------------------------------------
__device__ __forceinline__ void stage_weight_T(__bf16* sh_w,
                                               const float* __restrict__ W,
                                               int tid) {
  // W row-major [k][n] -> sh_w N-major [n][k] (bf16) for load_B.
  for (int i = tid; i < DIM * DIM; i += TPB1) {
    int k = i >> 7, n = i & 127;
    sh_w[n * DIM + k] = (__bf16)W[i];
  }
}

__global__ __launch_bounds__(TPB1) void fan_attn_kernel(
    const float* __restrict__ x,
    const float* __restrict__ scalar_W, const float* __restrict__ scalar_b,
    const float* __restrict__ hist_W,   const float* __restrict__ hist_b,
    const float* __restrict__ qW, const float* __restrict__ qb,
    const float* __restrict__ kW, const float* __restrict__ kb,
    const float* __restrict__ vW, const float* __restrict__ vb,
    const float* __restrict__ ln_g, const float* __restrict__ ln_b,
    const float* __restrict__ log_kern,
    __bf16* __restrict__ comb) {
  __shared__ __bf16 sh_h[SEQ * DIM];     // 64KB  residual stream, row-major
  __shared__ __bf16 sh_k[SEQ * DIM];     // 64KB  K, row-major [t][d]
  __shared__ __bf16 sh_vt[DIM * SEQ];    // 64KB  V transposed [d][t]
  __shared__ __bf16 sh_w[DIM * DIM];     // 32KB  staged weight, N-major
  __shared__ __bf16 sh_q[16 * DIM];      // 4KB   current Q tile
  __shared__ __bf16 sh_attn[16 * SEQ];   // 8KB   exp(scores) tile (unnorm.)
  __shared__ float  sh_out[16 * DIM];    // 8KB   attention-out tile (f32)
  __shared__ float  sh_pmax[8][16];
  __shared__ float  sh_psum[8][16];
  __shared__ float  sh_rmax[16];
  __shared__ float  sh_rinv[16];
  __shared__ float  sh_ln1[16][16];
  __shared__ float  sh_ln2[16][16];
  __shared__ float  sh_mu[16];
  __shared__ float  sh_rs[16];

  const int b    = blockIdx.x;
  const int tid  = threadIdx.x;
  const int wave = tid >> 5;
  const int lane = tid & 31;
  const int N    = lane & 15;            // C/B fragment column
  const int Mb   = (lane >> 4) << 3;     // C fragment row base
  const float inv_sqrt_d = 0.08838834764831845f;  // 1/sqrt(128)

  // ---- embed: h[s,d] = x_hist[s] * hist_W[d] + hist_b[d]  (b128 stores) ----
  for (int c = tid; c < SEQ * DIM / 8; c += TPB1) {
    int s = c >> 4, d0 = (c & 15) * 8;
    float xs = x[b * (NSC + SEQ) + NSC + s];
    __bf16 tmp[8] __attribute__((aligned(16)));
    #pragma unroll
    for (int j = 0; j < 8; ++j)
      tmp[j] = (__bf16)(xs * hist_W[d0 + j] + hist_b[d0 + j]);
    *(uint4*)&sh_h[s * DIM + d0] = *(uint4*)tmp;
  }
  __syncthreads();

  for (int layer = 0; layer < NLAY; ++layer) {
    const float* qWl = qW + layer * DIM * DIM;
    const float* kWl = kW + layer * DIM * DIM;
    const float* vWl = vW + layer * DIM * DIM;
    const float* qbl = qb + layer * DIM;
    const float* kbl = kb + layer * DIM;
    const float* vbl = vb + layer * DIM;
    const float* gl  = ln_g + layer * DIM;
    const float* bl  = ln_b + layer * DIM;

    // ---- K = h @ kW + kb  (row-major into sh_k) ----
    stage_weight_T(sh_w, kWl, tid);
    __syncthreads();
    for (int idx = wave; idx < 128; idx += 8) {
      int st = idx >> 3, nt = idx & 7;
      v8f c = {0.f,0.f,0.f,0.f,0.f,0.f,0.f,0.f};
      #pragma unroll
      for (int k0 = 0; k0 < DIM; k0 += 32)
        c = wmma_bf16(load_A(sh_h, DIM, st * 16, k0, lane),
                      load_B(sh_w, DIM, nt * 16, k0, lane), c);
      float bias = kbl[nt * 16 + N];
      #pragma unroll
      for (int j = 0; j < 8; ++j)
        sh_k[(st * 16 + Mb + j) * DIM + nt * 16 + N] = (__bf16)(c[j] + bias);
    }
    __syncthreads();

    // ---- V = h @ vW + vb  (transposed into sh_vt, packed b128 stores) ----
    stage_weight_T(sh_w, vWl, tid);
    __syncthreads();
    for (int idx = wave; idx < 128; idx += 8) {
      int st = idx >> 3, nt = idx & 7;
      v8f c = {0.f,0.f,0.f,0.f,0.f,0.f,0.f,0.f};
      #pragma unroll
      for (int k0 = 0; k0 < DIM; k0 += 32)
        c = wmma_bf16(load_A(sh_h, DIM, st * 16, k0, lane),
                      load_B(sh_w, DIM, nt * 16, k0, lane), c);
      float bias = vbl[nt * 16 + N];
      __bf16 tmp[8] __attribute__((aligned(16)));
      #pragma unroll
      for (int j = 0; j < 8; ++j) tmp[j] = (__bf16)(c[j] + bias);
      *(uint4*)&sh_vt[(nt * 16 + N) * SEQ + st * 16 + Mb] = *(uint4*)tmp;
    }
    __syncthreads();

    // ---- stage qW once; stream Q in 16-row tiles (flash-style) ----
    stage_weight_T(sh_w, qWl, tid);
    __syncthreads();

    for (int qi = 0; qi < 16; ++qi) {
      const int s0 = qi * 16;

      // prefetch the 16 log_kern rows this tile will read (4 lines each)
      if (tid < 64)
        __builtin_prefetch(&log_kern[(s0 + (tid >> 2)) * SEQ + (tid & 3) * 64],
                           0, 3);

      // Q tile = h[s0:s0+16] @ qW + qb  (one 16-col tile per wave)
      {
        int nt = wave;
        v8f c = {0.f,0.f,0.f,0.f,0.f,0.f,0.f,0.f};
        #pragma unroll
        for (int k0 = 0; k0 < DIM; k0 += 32)
          c = wmma_bf16(load_A(sh_h, DIM, s0, k0, lane),
                        load_B(sh_w, DIM, nt * 16, k0, lane), c);
        float bias = qbl[nt * 16 + N];
        #pragma unroll
        for (int j = 0; j < 8; ++j)
          sh_q[(Mb + j) * DIM + nt * 16 + N] = (__bf16)(c[j] + bias);
      }
      __syncthreads();

      // scores tile: 16 x 256; wave w owns column tiles 2w, 2w+1
      v8f sc[2];
      #pragma unroll
      for (int t = 0; t < 2; ++t) {
        int ct = wave * 2 + t;
        v8f c = {0.f,0.f,0.f,0.f,0.f,0.f,0.f,0.f};
        #pragma unroll
        for (int k0 = 0; k0 < DIM; k0 += 32)
          c = wmma_bf16(load_A(sh_q, DIM, 0, k0, lane),
                        load_B(sh_k, DIM, ct * 16, k0, lane), c);
        #pragma unroll
        for (int j = 0; j < 8; ++j)
          c[j] = c[j] * inv_sqrt_d +
                 log_kern[(s0 + Mb + j) * SEQ + ct * 16 + N];
        sc[t] = c;
      }

      // row max: in-lane over 2 tiles, half-wave shuffle, cross-wave via LDS
      float mj[8];
      #pragma unroll
      for (int j = 0; j < 8; ++j) {
        mj[j] = fmaxf(sc[0][j], sc[1][j]);
        #pragma unroll
        for (int off = 8; off >= 1; off >>= 1)
          mj[j] = fmaxf(mj[j], __shfl_xor(mj[j], off, 32));
      }
      if (N == 0) {
        #pragma unroll
        for (int j = 0; j < 8; ++j) sh_pmax[wave][Mb + j] = mj[j];
      }
      __syncthreads();
      if (tid < 16) {
        float m = sh_pmax[0][tid];
        #pragma unroll
        for (int w = 1; w < 8; ++w) m = fmaxf(m, sh_pmax[w][tid]);
        sh_rmax[tid] = m;
      }
      __syncthreads();

      // p = exp(score - rowmax) -> bf16 attn tile; accumulate row sums
      float sj[8] = {0.f,0.f,0.f,0.f,0.f,0.f,0.f,0.f};
      #pragma unroll
      for (int t = 0; t < 2; ++t) {
        int ct = wave * 2 + t;
        #pragma unroll
        for (int j = 0; j < 8; ++j) {
          int M = Mb + j;
          float p = __expf(sc[t][j] - sh_rmax[M]);
          sj[j] += p;
          sh_attn[M * SEQ + ct * 16 + N] = (__bf16)p;
        }
      }
      #pragma unroll
      for (int j = 0; j < 8; ++j) {
        #pragma unroll
        for (int off = 8; off >= 1; off >>= 1)
          sj[j] += __shfl_xor(sj[j], off, 32);
      }
      if (N == 0) {
        #pragma unroll
        for (int j = 0; j < 8; ++j) sh_psum[wave][Mb + j] = sj[j];
      }
      __syncthreads();
      if (tid < 16) {
        float s = 0.f;
        #pragma unroll
        for (int w = 0; w < 8; ++w) s += sh_psum[w][tid];
        sh_rinv[tid] = 1.f / s;
      }
      __syncthreads();

      // out tile = attn @ V  (one 16-col tile per wave)
      {
        int nt = wave;
        v8f c = {0.f,0.f,0.f,0.f,0.f,0.f,0.f,0.f};
        #pragma unroll
        for (int t0 = 0; t0 < SEQ; t0 += 32)
          c = wmma_bf16(load_A(sh_attn, SEQ, 0, t0, lane),
                        load_B(sh_vt, SEQ, nt * 16, t0, lane), c);
        #pragma unroll
        for (int j = 0; j < 8; ++j)
          sh_out[(Mb + j) * DIM + nt * 16 + N] = c[j] * sh_rinv[Mb + j];
      }
      __syncthreads();

      // residual + layernorm over D, write back to sh_h (bf16)
      {
        int r = tid >> 4, p = tid & 15;
        float s1 = 0.f, s2 = 0.f;
        for (int d = p * 8; d < p * 8 + 8; ++d) {
          float val = (float)sh_h[(s0 + r) * DIM + d] + sh_out[r * DIM + d];
          sh_out[r * DIM + d] = val;
          s1 += val; s2 += val * val;
        }
        sh_ln1[r][p] = s1; sh_ln2[r][p] = s2;
      }
      __syncthreads();
      if (tid < 16) {
        float s1 = 0.f, s2 = 0.f;
        #pragma unroll
        for (int p = 0; p < 16; ++p) { s1 += sh_ln1[tid][p]; s2 += sh_ln2[tid][p]; }
        float mu = s1 * (1.f / DIM);
        float var = s2 * (1.f / DIM) - mu * mu;
        sh_mu[tid] = mu;
        sh_rs[tid] = rsqrtf(var + 1e-5f);
      }
      __syncthreads();
      {
        int r = tid >> 4, p = tid & 15;
        float mu = sh_mu[r], rs = sh_rs[r];
        for (int d = p * 8; d < p * 8 + 8; ++d)
          sh_h[(s0 + r) * DIM + d] =
              (__bf16)((sh_out[r * DIM + d] - mu) * rs * gl[d] + bl[d]);
      }
      __syncthreads();
    }  // qi
  }    // layer

  // ---- emit combined row: [scalar_emb(128) | h(256*128)] as bf16 ----
  __bf16* crow = comb + (size_t)b * COMBW;
  if (tid < DIM) {
    float acc = scalar_b[tid];
    #pragma unroll
    for (int j = 0; j < NSC; ++j)
      acc += x[b * (NSC + SEQ) + j] * scalar_W[j * DIM + tid];
    crow[tid] = (__bf16)acc;
  }
  for (int c = tid; c < SEQ * DIM / 8; c += TPB1)
    ((uint4*)(crow + DIM))[c] = ((const uint4*)sh_h)[c];
}

// ---------------------------------------------------------------------------
// Kernel 2: z1 = relu(combined @ h1_W + h1_b)   (2048x32896)@(32896x128)
// 256 threads = 8 waves; each wave owns a 16x128 output strip.
// Double-buffered LDS slab of the bf16 weight image; raw-copy staging uses
// the CDNA5 async global->LDS path (ASYNCcnt) when available.
// ---------------------------------------------------------------------------
__device__ __forceinline__ void stage_slab(__bf16* dst,
                                           const __bf16* __restrict__ wt,
                                           int k0, int tid) {
  // copy 128 rows x 32 bf16 (64B/row) = 8KB as 512 16B chunks
  for (int i = tid; i < 512; i += 256) {
    int n = i >> 2, part = (i & 3) * 8;
    const __bf16* src = wt + (size_t)n * COMBW + k0 + part;
    __bf16* d = dst + n * 32 + part;
#if defined(HAVE_ASYNC_LDS)
    __builtin_amdgcn_global_load_async_to_lds_b128(
        (__attribute__((address_space(1))) v4i_vs*)src,
        (__attribute__((address_space(3))) v4i_vs*)d, 0, 0);
#else
    *(uint4*)d = *(const uint4*)src;
#endif
    __builtin_prefetch(src + 64, 0, 1);   // next-next slab into L2
  }
}

__global__ __launch_bounds__(256) void fan_mlp1_kernel(
    const __bf16* __restrict__ comb,
    const __bf16* __restrict__ wt, const float* __restrict__ h1_b,
    float* __restrict__ z1) {
  __shared__ __bf16 sh_bt[2][DIM * 32];   // 2 x 8KB k-slabs, N-major [n][k']
  const int tid  = threadIdx.x;
  const int wave = tid >> 5;
  const int lane = tid & 31;
  const int N    = lane & 15;
  const int Mb   = (lane >> 4) << 3;
  const int row0 = blockIdx.x * 128 + wave * 16;

  v8f acc[8];
  #pragma unroll
  for (int nt = 0; nt < 8; ++nt)
    acc[nt] = (v8f){0.f,0.f,0.f,0.f,0.f,0.f,0.f,0.f};

  stage_slab(sh_bt[0], wt, 0, tid);
  int cur = 0;
  for (int k0 = 0; k0 < COMBW; k0 += 32) {
#if defined(HAVE_ASYNC_LDS)
    asm volatile("s_wait_asynccnt 0x0" ::: "memory");
#endif
    __syncthreads();
    if (k0 + 32 < COMBW) stage_slab(sh_bt[cur ^ 1], wt, k0 + 32, tid);
    v16bf a = load_A(comb, COMBW, row0, k0, lane);
    #pragma unroll
    for (int nt = 0; nt < 8; ++nt)
      acc[nt] = wmma_bf16(a, load_B(sh_bt[cur], 32, nt * 16, 0, lane), acc[nt]);
    cur ^= 1;
  }
  #pragma unroll
  for (int nt = 0; nt < 8; ++nt) {
    float bias = h1_b[nt * 16 + N];
    #pragma unroll
    for (int j = 0; j < 8; ++j) {
      float v = acc[nt][j] + bias;
      z1[(size_t)(row0 + Mb + j) * DIM + nt * 16 + N] = fmaxf(v, 0.f);
    }
  }
}

// ---------------------------------------------------------------------------
// Kernel 3: z2 = relu(z1 @ h2_W + h2_b); out = z2 @ h3_W + h3_b
// One block per batch element, 64 threads (2 waves).
// ---------------------------------------------------------------------------
__global__ __launch_bounds__(64) void fan_mlp2_kernel(
    const float* __restrict__ z1,
    const float* __restrict__ h2_W, const float* __restrict__ h2_b,
    const float* __restrict__ h3_W, const float* __restrict__ h3_b,
    float* __restrict__ out) {
  __shared__ float red[2];
  const int b = blockIdx.x;
  const int j = threadIdx.x;
  const float* zr = z1 + (size_t)b * DIM;
  float acc = h2_b[j];
  for (int k = 0; k < DIM; ++k) acc += zr[k] * h2_W[k * 64 + j];
  acc = fmaxf(acc, 0.f) * h3_W[j];
  #pragma unroll
  for (int off = 16; off >= 1; off >>= 1) acc += __shfl_xor(acc, off, 32);
  if ((j & 31) == 0) red[j >> 5] = acc;
  __syncthreads();
  if (j == 0) out[b] = red[0] + red[1] + h3_b[0];
}

// ---------------------------------------------------------------------------
extern "C" void kernel_launch(void* const* d_in, const int* in_sizes, int n_in,
                              void* d_out, int out_size, void* d_ws,
                              size_t ws_size, hipStream_t stream) {
  const float* x        = (const float*)d_in[0];
  const float* scalar_W = (const float*)d_in[1];
  const float* scalar_b = (const float*)d_in[2];
  const float* hist_W   = (const float*)d_in[3];
  const float* hist_b   = (const float*)d_in[4];
  const float* qW       = (const float*)d_in[5];
  const float* qb       = (const float*)d_in[6];
  const float* kW       = (const float*)d_in[7];
  const float* kb       = (const float*)d_in[8];
  const float* vW       = (const float*)d_in[9];
  const float* vb       = (const float*)d_in[10];
  const float* ln_g     = (const float*)d_in[11];
  const float* ln_b     = (const float*)d_in[12];
  const float* h1_W     = (const float*)d_in[13];
  const float* h1_b     = (const float*)d_in[14];
  const float* h2_W     = (const float*)d_in[15];
  const float* h2_b     = (const float*)d_in[16];
  const float* h3_W     = (const float*)d_in[17];
  const float* h3_b     = (const float*)d_in[18];
  float* out = (float*)d_out;

  char* ws = (char*)d_ws;
  const size_t LK_OFF   = 0;
  const size_t COMB_OFF = LK_OFF + (size_t)SEQ * SEQ * sizeof(float);
  const size_t Z1_OFF   = COMB_OFF + (size_t)BATCH * COMBW * sizeof(__bf16);
  const size_t WT_OFF   = Z1_OFF + (size_t)BATCH * DIM * sizeof(float);
  float*  lk   = (float*)(ws + LK_OFF);
  __bf16* comb = (__bf16*)(ws + COMB_OFF);
  float*  z1   = (float*)(ws + Z1_OFF);
  __bf16* wt   = (__bf16*)(ws + WT_OFF);

  fan_logkern_kernel<<<1, 256, 0, stream>>>(lk);
  {
    dim3 g((COMBW + 255) / 256, DIM);
    fan_wconv_kernel<<<g, 256, 0, stream>>>(h1_W, wt);
  }
  fan_attn_kernel<<<BATCH, TPB1, 0, stream>>>(
      x, scalar_W, scalar_b, hist_W, hist_b, qW, qb, kW, kb, vW, vb,
      ln_g, ln_b, lk, comb);
  fan_mlp1_kernel<<<BATCH / 128, 256, 0, stream>>>(comb, wt, h1_b, z1);
  fan_mlp2_kernel<<<BATCH, 64, 0, stream>>>(z1, h2_W, h2_b, h3_W, h3_b, out);
}
